// MultiHeadAttention_73203422593774
// MI455X (gfx1250) — matrix operand order repaired and checked by reading, not compiled, not run on hardware
//
#include <hip/hip_runtime.h>

// ---------------------------------------------------------------------------
// MI455X (gfx1250) multi-head attention: fp32 in -> bf16 WMMA -> fp32 out
// B=2, S=2048, D=1024, H=16, Dh=64, causal, scale = 1/sqrt(1024) = 1/32
// Pipeline: cast/transpose -> QKV GEMMs (V stored transposed) -> flash attn
//           -> output GEMM.  LDS staging via Tensor Data Mover when available.
// ---------------------------------------------------------------------------

typedef __attribute__((ext_vector_type(16))) __bf16 v16bf;
typedef __attribute__((ext_vector_type(8)))  __bf16 v8bf;
typedef __attribute__((ext_vector_type(8)))  float  v8f;
typedef __attribute__((ext_vector_type(4)))  unsigned int u32x4;
typedef __attribute__((ext_vector_type(8)))  int i32x8;
typedef __attribute__((ext_vector_type(4)))  int i32x4;

#define DEV __device__ __forceinline__

static constexpr int BATCH  = 2;
static constexpr int SEQ    = 2048;
static constexpr int DMODEL = 1024;
static constexpr int HEADS  = 16;
static constexpr int DH     = 64;
static constexpr int MROWS  = BATCH * SEQ;       // 4096
static constexpr float SM_SCALE = 0.03125f;      // 1/sqrt(1024)

// --- Tensor Data Mover availability (device pass only; host sees 0) --------
#if defined(__has_builtin)
#if __has_builtin(__builtin_amdgcn_tensor_load_to_lds) && \
    __has_builtin(__builtin_amdgcn_s_wait_tensorcnt)
#define HAVE_TDM 1
#endif
#endif
#ifndef HAVE_TDM
#define HAVE_TDM 0
#endif

DEV v16bf cat16(v8bf lo, v8bf hi) {
  return __builtin_shufflevector(lo, hi, 0,1,2,3,4,5,6,7,8,9,10,11,12,13,14,15);
}

DEV v8f wmma_bf16(v16bf a, v16bf b, v8f c) {
  return __builtin_amdgcn_wmma_f32_16x16x32_bf16(
      false, a, false, b, (short)0, c, false, false);
}

DEV float redmax16(float v) {
  v = fmaxf(v, __shfl_xor(v, 1, 32));
  v = fmaxf(v, __shfl_xor(v, 2, 32));
  v = fmaxf(v, __shfl_xor(v, 4, 32));
  v = fmaxf(v, __shfl_xor(v, 8, 32));
  return v;
}
DEV float redsum16(float v) {
  v += __shfl_xor(v, 1, 32);
  v += __shfl_xor(v, 2, 32);
  v += __shfl_xor(v, 4, 32);
  v += __shfl_xor(v, 8, 32);
  return v;
}

// ---------------------------------------------------------------------------
// TDM 2-D tile load: global (row-major, stride_elems) -> LDS (tile row-major).
// D# built per CDNA5 ISA ch.8: group0 {count,lds_addr,global_addr,type=2},
// group1 {data_size=2B, tensor dims (huge: no OOB), tile dims, dim0 stride}.
// ---------------------------------------------------------------------------
DEV void tdm_load_2d(const void* lds_ptr, const void* gptr,
                     unsigned tile_w, unsigned tile_h, unsigned stride_elems) {
#if HAVE_TDM
  unsigned long long ga = (unsigned long long)gptr;
  unsigned lds_addr = (unsigned)(unsigned long long)lds_ptr;  // LDS aperture: low 32 bits
  u32x4 g0;
  g0[0] = 1u;                                        // count=1, user descriptor
  g0[1] = lds_addr;                                  // [63:32] lds_addr
  g0[2] = (unsigned)(ga & 0xffffffffu);              // [95:64] global_addr lo
  g0[3] = (unsigned)((ga >> 32) & 0x01ffffffu)       // [120:96] global_addr hi
        | (2u << 30);                                // [127:126] type=2 (image)
  const unsigned td0 = 0x40000000u, td1 = 0x40000000u;  // huge -> no OOB clamp
  i32x8 g1;
  g1[0] = (int)(1u << 16);                           // data_size=1 (2 bytes)
  g1[1] = (int)((td0 & 0xffffu) << 16);              // [63:48]=tensor_dim0 lo
  g1[2] = (int)(((td0 >> 16) & 0xffffu) | ((td1 & 0xffffu) << 16));
  g1[3] = (int)(((td1 >> 16) & 0xffffu) | (tile_w << 16));   // tile_dim0
  g1[4] = (int)(tile_h & 0xffffu);                   // tile_dim1 (tile_dim2=0)
  g1[5] = (int)stride_elems;                         // tensor_dim0_stride lo
  g1[6] = 0;
  g1[7] = 0;
  i32x4 z4 = {};
#if __clang_major__ >= 23
  i32x8 z8 = {};
  __builtin_amdgcn_tensor_load_to_lds(g0, g1, z4, z4, z8, 0);
#else
  __builtin_amdgcn_tensor_load_to_lds(g0, g1, z4, z4, 0);
#endif
#else
  (void)lds_ptr; (void)gptr; (void)tile_w; (void)tile_h; (void)stride_elems;
#endif
}

DEV void tdm_wait() {
#if HAVE_TDM
  __builtin_amdgcn_s_wait_tensorcnt(0);
#endif
}

// ---------------------------------------------------------------------------
// fp32 -> bf16 cast (x activations)
// ---------------------------------------------------------------------------
__global__ void __launch_bounds__(256)
cast_f32_to_bf16(const float* __restrict__ in, __bf16* __restrict__ out, int n) {
  int i = (blockIdx.x * blockDim.x + threadIdx.x) * 4;
  if (i + 3 < n) {
    float4 f = *(const float4*)(in + i);
    out[i + 0] = (__bf16)f.x;
    out[i + 1] = (__bf16)f.y;
    out[i + 2] = (__bf16)f.z;
    out[i + 3] = (__bf16)f.w;
  }
}

// ---------------------------------------------------------------------------
// fp32 [1024][1024] -> bf16 transposed [1024][1024] (weights, done once)
// ---------------------------------------------------------------------------
__global__ void __launch_bounds__(256)
transpose_cast_1024(const float* __restrict__ in, __bf16* __restrict__ out) {
  __shared__ __bf16 tile[64][64];
  const int n0 = blockIdx.x * 64, k0 = blockIdx.y * 64;
  const int t = threadIdx.x;
  #pragma unroll
  for (int r = 0; r < 4; ++r) {
    int idx = t + r * 256;           // 1024 float4 loads
    int row = idx >> 4;              // 0..63 (k)
    int c4  = idx & 15;
    float4 f = *(const float4*)(&in[(size_t)(k0 + row) * DMODEL + n0 + c4 * 4]);
    tile[row][c4 * 4 + 0] = (__bf16)f.x;
    tile[row][c4 * 4 + 1] = (__bf16)f.y;
    tile[row][c4 * 4 + 2] = (__bf16)f.z;
    tile[row][c4 * 4 + 3] = (__bf16)f.w;
  }
  __syncthreads();
  #pragma unroll
  for (int r = 0; r < 2; ++r) {
    int idx  = t + r * 256;          // 512 uint4 stores
    int nrow = idx >> 3;             // 0..63 (n)
    int c8   = idx & 7;
    __bf16 v[8];
    #pragma unroll
    for (int e = 0; e < 8; ++e) v[e] = tile[c8 * 8 + e][nrow];
    *(uint4*)(&out[(size_t)(n0 + nrow) * DMODEL + k0 + c8 * 8]) =
        *(const uint4*)v;
  }
}

// ---------------------------------------------------------------------------
// GEMM: C[4096x1024] = A[4096x1024] * Bt[1024x1024]^T (Bt is N-major).
// Block: 128 thr (4 waves), tile 128x64, BK=32; wave -> 32 rows (8 WMMA/chunk).
// MODE: 0 = bf16 row-major, 1 = f32 row-major, 2 = bf16 V-transposed
//       ([B*H*Dh][SEQ], packed b128 stores along tokens).
// ---------------------------------------------------------------------------
template <int MODE>
__global__ void __launch_bounds__(128)
gemm_bt(const __bf16* __restrict__ A, const __bf16* __restrict__ Bt,
        void* __restrict__ Cout) {
  constexpr int K = DMODEL, N = DMODEL;
  constexpr int BM = 128, BN = 64, BK = 32;
  __shared__ __bf16 As[2][BM * BK];    // [row][k]
  __shared__ __bf16 Bs[2][BN * BK];    // [n][k]  (Bt rows copy straight in)

  const int n0 = blockIdx.x * BN;
  const int m0 = blockIdx.y * BM;
  const int t    = threadIdx.x;
  const int wave = t >> 5;
  const int lane = t & 31;
  const int l16  = lane & 15;
  const int hh   = lane >> 4;

  v8f acc[2][4] = {};

  auto compute = [&](int buf) {
    v16bf af[2];
    #pragma unroll
    for (int rt = 0; rt < 2; ++rt) {
      const __bf16* arow = &As[buf][(wave * 32 + rt * 16 + l16) * BK];
      af[rt] = cat16(*(const v8bf*)(arow + hh * 16),
                     *(const v8bf*)(arow + hh * 16 + 8));
    }
    #pragma unroll
    for (int nt = 0; nt < 4; ++nt) {
      const __bf16* bcol = &Bs[buf][(nt * 16 + l16) * BK];
      v16bf bf = cat16(*(const v8bf*)(bcol + hh * 16),
                       *(const v8bf*)(bcol + hh * 16 + 8));
      #pragma unroll
      for (int rt = 0; rt < 2; ++rt)
        acc[rt][nt] = wmma_bf16(af[rt], bf, acc[rt][nt]);
    }
  };

#if HAVE_TDM
  auto stage = [&](int kc, int buf) {
    tdm_load_2d(&As[buf][0], &A[(size_t)m0 * K + kc], BK, BM, K);
    tdm_load_2d(&Bs[buf][0], &Bt[(size_t)n0 * K + kc], BK, BN, K);
  };
  if (wave == 0) stage(0, 0);
  for (int kc = 0, it = 0; kc < K; kc += BK, ++it) {
    const int buf = it & 1;
    if (wave == 0) tdm_wait();
    __syncthreads();
    if (wave == 0 && kc + BK < K) stage(kc + BK, buf ^ 1);  // prefetch overlap
    compute(buf);
    __syncthreads();
  }
#else
  for (int kc = 0; kc < K; kc += BK) {
    #pragma unroll
    for (int r = 0; r < 4; ++r) {        // A: 128x32 = 512 uint4
      int idx = t + r * 128;
      int row = idx >> 2, c4 = idx & 3;
      *(uint4*)(&As[0][row * BK + c4 * 8]) =
          *(const uint4*)(&A[(size_t)(m0 + row) * K + kc + c4 * 8]);
    }
    #pragma unroll
    for (int r = 0; r < 2; ++r) {        // B: 64x32 = 256 uint4
      int idx = t + r * 128;
      int row = idx >> 2, c4 = idx & 3;
      *(uint4*)(&Bs[0][row * BK + c4 * 8]) =
          *(const uint4*)(&Bt[(size_t)(n0 + row) * K + kc + c4 * 8]);
    }
    __syncthreads();
    compute(0);
    __syncthreads();
  }
#endif

  // ---- epilogue: C/D element j -> row hh*8+j, col l16
  #pragma unroll
  for (int rt = 0; rt < 2; ++rt) {
    #pragma unroll
    for (int nt = 0; nt < 4; ++nt) {
      const int row0 = m0 + wave * 32 + rt * 16 + hh * 8;
      const int col  = n0 + nt * 16 + l16;
      if (MODE == 1) {
        #pragma unroll
        for (int j = 0; j < 8; ++j)
          ((float*)Cout)[(size_t)(row0 + j) * N + col] = acc[rt][nt][j];
      } else if (MODE == 0) {
        #pragma unroll
        for (int j = 0; j < 8; ++j)
          ((__bf16*)Cout)[(size_t)(row0 + j) * N + col] = (__bf16)acc[rt][nt][j];
      } else {  // MODE == 2: V^T, tokens contiguous per lane -> packed b128
        __bf16 hb[8];
        #pragma unroll
        for (int j = 0; j < 8; ++j) hb[j] = (__bf16)acc[rt][nt][j];
        const int b = row0 >> 11, s = row0 & (SEQ - 1);
        const int head = col >> 6, dh = col & (DH - 1);
        __bf16* dst = (__bf16*)Cout +
            ((((size_t)b * HEADS + head) * DH + dh) * SEQ + s);
        *(uint4*)dst = *(const uint4*)hb;
      }
    }
  }
}

// ---------------------------------------------------------------------------
// Causal flash attention. Grid (S/64, H, B), block 128 thr = 4 waves.
// Q row-major [tok][D]; K row-major [tok][D]; Vt [(b*H+h)*Dh][SEQ].
// ---------------------------------------------------------------------------
__global__ void __launch_bounds__(128)
attn_kernel(const __bf16* __restrict__ Q, const __bf16* __restrict__ Km,
            const __bf16* __restrict__ Vt, __bf16* __restrict__ ctx) {
  __shared__ __bf16 Ks [2][64 * 64];    // [key][dh]  (N-major for QK^T B-frag)
  __shared__ __bf16 Vts[2][64 * 64];    // [dh][key]  (N-major for PV  B-frag)
  __shared__ __bf16 Ps [4][16 * 64];    // per-wave P tile [qrow][key]

  const int qt = blockIdx.x, h = blockIdx.y, b = blockIdx.z;
  const int t    = threadIdx.x;
  const int wave = t >> 5;
  const int lane = t & 31;
  const int l16  = lane & 15;
  const int hh   = lane >> 4;

  const size_t hoff = (size_t)h * DH;
  const __bf16* Qbase  = Q  + ((size_t)b * SEQ) * DMODEL + hoff;
  const __bf16* Kbase  = Km + ((size_t)b * SEQ) * DMODEL + hoff;
  const __bf16* Vtbase = Vt + ((size_t)(b * HEADS + h) * DH) * SEQ;
  const int q0 = qt * 64 + wave * 16;

  // Q fragments in registers: Dh=64 -> 2 chunks of K=32
  v16bf qf[2];
  #pragma unroll
  for (int kc = 0; kc < 2; ++kc) {
    const __bf16* rowp = Qbase + (size_t)(q0 + l16) * DMODEL + kc * 32;
    qf[kc] = cat16(*(const v8bf*)(rowp + hh * 16),
                   *(const v8bf*)(rowp + hh * 16 + 8));
  }

  v8f o[4] = {};
  float mj[8], lj[8];
  #pragma unroll
  for (int j = 0; j < 8; ++j) { mj[j] = -1e30f; lj[j] = 0.f; }

#if HAVE_TDM
  auto stage = [&](int kt, int buf) {
    tdm_load_2d(&Ks[buf][0],  Kbase  + (size_t)kt * 64 * DMODEL, 64, 64, DMODEL);
    tdm_load_2d(&Vts[buf][0], Vtbase + (size_t)kt * 64,          64, 64, SEQ);
  };
  if (wave == 0) stage(0, 0);
#endif

  for (int kt = 0; kt <= qt; ++kt) {
    const int buf = kt & 1;
#if HAVE_TDM
    if (wave == 0) tdm_wait();
    __syncthreads();
    if (wave == 0 && kt + 1 <= qt) stage(kt + 1, buf ^ 1);  // prefetch overlap
#else
    #pragma unroll
    for (int r = 0; r < 4; ++r) {        // 512 uint4 each, straight copies
      int idx = t + r * 128;
      int row = idx >> 3, c8 = idx & 7;
      *(uint4*)(&Ks[buf][row * 64 + c8 * 8]) = *(const uint4*)(
          &Kbase[((size_t)kt * 64 + row) * DMODEL + c8 * 8]);
      *(uint4*)(&Vts[buf][row * 64 + c8 * 8]) = *(const uint4*)(
          &Vtbase[(size_t)row * SEQ + kt * 64 + c8 * 8]);
    }
    __syncthreads();
#endif

    // ---- scores: 16 q rows x 64 keys
    v8f s[4] = {};
    #pragma unroll
    for (int nk = 0; nk < 4; ++nk) {
      #pragma unroll
      for (int kc = 0; kc < 2; ++kc) {
        const __bf16* kp = &Ks[buf][(nk * 16 + l16) * 64 + kc * 32 + hh * 16];
        v16bf bfrag = cat16(*(const v8bf*)kp, *(const v8bf*)(kp + 8));
        s[nk] = wmma_bf16(qf[kc], bfrag, s[nk]);
      }
    }

    // ---- scale + causal mask (diagonal tile only)
    const bool diag = (kt == qt);
    #pragma unroll
    for (int nk = 0; nk < 4; ++nk) {
      #pragma unroll
      for (int j = 0; j < 8; ++j) {
        float x = s[nk][j] * SM_SCALE;
        if (diag) {
          int qrow = q0 + hh * 8 + j;
          int key  = kt * 64 + nk * 16 + l16;
          if (key > qrow) x = -1e30f;
        }
        s[nk][j] = x;
      }
    }

    // ---- online softmax
    #pragma unroll
    for (int j = 0; j < 8; ++j) {
      float mn = mj[j];
      #pragma unroll
      for (int nk = 0; nk < 4; ++nk) mn = fmaxf(mn, s[nk][j]);
      mn = redmax16(mn);
      float alpha = __expf(mj[j] - mn);
      float rs = 0.f;
      #pragma unroll
      for (int nk = 0; nk < 4; ++nk) {
        float p = __expf(s[nk][j] - mn);
        s[nk][j] = p;
        rs += p;
      }
      rs = redsum16(rs);
      lj[j] = lj[j] * alpha + rs;
      mj[j] = mn;
      #pragma unroll
      for (int dt = 0; dt < 4; ++dt) o[dt][j] *= alpha;
      #pragma unroll
      for (int nk = 0; nk < 4; ++nk)
        Ps[wave][(hh * 8 + j) * 64 + nk * 16 + l16] = (__bf16)s[nk][j];
    }

    // ---- PV accumulate (per-wave LDS, DS ops in-order within wave)
    #pragma unroll
    for (int kc = 0; kc < 2; ++kc) {
      const __bf16* pp = &Ps[wave][l16 * 64 + kc * 32];
      v16bf pf = cat16(*(const v8bf*)(pp + hh * 16),
                       *(const v8bf*)(pp + hh * 16 + 8));
      #pragma unroll
      for (int dt = 0; dt < 4; ++dt) {
        const __bf16* vp = &Vts[buf][(dt * 16 + l16) * 64 + kc * 32 + hh * 16];
        v16bf vf = cat16(*(const v8bf*)vp, *(const v8bf*)(vp + 8));
        o[dt] = wmma_bf16(pf, vf, o[dt]);
      }
    }
    __syncthreads();
  }

  // ---- normalize + store ctx bf16 [B*S][D]
  float inv[8];
  #pragma unroll
  for (int j = 0; j < 8; ++j) inv[j] = 1.f / lj[j];
  #pragma unroll
  for (int dt = 0; dt < 4; ++dt) {
    #pragma unroll
    for (int j = 0; j < 8; ++j) {
      int qrow = q0 + hh * 8 + j;
      ctx[((size_t)b * SEQ + qrow) * DMODEL + hoff + dt * 16 + l16] =
          (__bf16)(o[dt][j] * inv[j]);
    }
  }
}

// ---------------------------------------------------------------------------
// Host orchestration
// ---------------------------------------------------------------------------
extern "C" void kernel_launch(void* const* d_in, const int* in_sizes, int n_in,
                              void* d_out, int out_size, void* d_ws, size_t ws_size,
                              hipStream_t stream) {
  const float* x  = (const float*)d_in[0];
  const float* Wq = (const float*)d_in[1];
  const float* Wk = (const float*)d_in[2];
  const float* Wv = (const float*)d_in[3];
  const float* Wo = (const float*)d_in[4];
  float* out = (float*)d_out;

  const size_t SZ_X = (size_t)MROWS * DMODEL;   // 4M elems
  const size_t SZ_W = (size_t)DMODEL * DMODEL;  // 1M elems

  __bf16* xb  = (__bf16*)d_ws;
  __bf16* wqt = xb  + SZ_X;
  __bf16* wkt = wqt + SZ_W;
  __bf16* wvt = wkt + SZ_W;
  __bf16* wot = wvt + SZ_W;
  __bf16* Qb  = wot + SZ_W;
  __bf16* Kb  = Qb  + SZ_X;
  __bf16* Vtb = Kb  + SZ_X;     // transposed V: [(b*H+h)*Dh][SEQ]
  __bf16* Cb  = Vtb + SZ_X;     // ctx

  // 1) casts / weight transposes (once, off the hot path)
  cast_f32_to_bf16<<<(int)(SZ_X / 1024), 256, 0, stream>>>(x, xb, (int)SZ_X);
  dim3 tgrid(16, 16);
  transpose_cast_1024<<<tgrid, 256, 0, stream>>>(Wq, wqt);
  transpose_cast_1024<<<tgrid, 256, 0, stream>>>(Wk, wkt);
  transpose_cast_1024<<<tgrid, 256, 0, stream>>>(Wv, wvt);
  transpose_cast_1024<<<tgrid, 256, 0, stream>>>(Wo, wot);

  // 2) projections (V stored transposed for attention's PV operand)
  dim3 ggrid(DMODEL / 64, MROWS / 128);
  gemm_bt<0><<<ggrid, 128, 0, stream>>>(xb, wqt, Qb);
  gemm_bt<0><<<ggrid, 128, 0, stream>>>(xb, wkt, Kb);
  gemm_bt<2><<<ggrid, 128, 0, stream>>>(xb, wvt, Vtb);

  // 3) causal flash attention
  dim3 agrid(SEQ / 64, HEADS, BATCH);
  attn_kernel<<<agrid, 128, 0, stream>>>(Qb, Kb, Vtb, Cb);

  // 4) output projection -> f32
  gemm_bt<1><<<ggrid, 128, 0, stream>>>(Cb, wot, out);
}
